// PrototypeInteractionBlock_61538291417487
// MI455X (gfx1250) — matrix operand order repaired and checked by reading, not compiled, not run on hardware
//
#include <hip/hip_runtime.h>
#include <hip/hip_bf16.h>
#include <math.h>

// ---------------------------------------------------------------------------
// PrototypeInteractionBlock fused pipeline for MI455X (gfx1250, wave32, WMMA)
//
//   K_pack   : pack q_w/out_w into WMMA B-fragment layout (bf16)
//   K_protokv: k = proto@k_w^T+k_b, v likewise; pack into B-frag layout
//   K_main   : pos-bilinear + Q GEMM + softmax(P=16) + attn@v + out-proj GEMM
//              (all GEMMs via v_wmma_f32_16x16x32_bf16), stores s in NCHW
//   K_local  : depthwise3x3 + BN + exact GELU + residuals -> y, per-(b,c) sums
//   K_gn     : GroupNorm(8) finalize -> d_out
// ---------------------------------------------------------------------------

typedef __attribute__((ext_vector_type(16))) __bf16        bf16x16;
typedef __attribute__((ext_vector_type(8)))  float         f32x8;
typedef __attribute__((ext_vector_type(4)))  unsigned int  u32x4;

#define NB   16
#define NC   256
#define NH   96
#define NW   96
#define NHW  9216
#define NP   16
#define NG   8

union FragU { u32x4 q[2]; bf16x16 v; };

__device__ __forceinline__ unsigned short f2bf(float f) {
  unsigned int u = __float_as_uint(f);
  u += 0x7FFFu + ((u >> 16) & 1u);          // round-to-nearest-even
  return (unsigned short)(u >> 16);
}
__device__ __forceinline__ f32x8 splat8(float x) {
  f32x8 r = {x, x, x, x, x, x, x, x};
  return r;
}

// ---------------- workspace layout (bytes) ----------------
// partial sums per (b,c): 4096 * {sum,sumsq}
static constexpr size_t WS_PS = 0;                         // 32768 B
static constexpr size_t WS_KB = 32768;                     // 8 frags  * 1KB
static constexpr size_t WS_VB = 40960;                     // 16 frags * 1KB
static constexpr size_t WS_QB = 57344;                     // 128 frags * 1KB
static constexpr size_t WS_OB = 188416;                    // 128 frags * 1KB
static constexpr size_t WS_S  = 327680;                    // 150994944 B (NCHW f32)
static constexpr size_t WS_Y  = WS_S + (size_t)150994944;  // 150994944 B (NCHW f32)

// ===========================================================================
// Pack q_w / out_w (row-major [d][c]) into B-fragment layout:
//   elem e = frag*512 + lane*16 + p ; frag = nt*8+kt
//   value  = W[n][K], n = nt*16 + (lane&15), K = kt*32 + (lane>=16?8:0) + (p&7) + (p>=8?16:0)
// ===========================================================================
__global__ void __launch_bounds__(256)
pib_pack_kernel(const float* __restrict__ q_w, const float* __restrict__ out_w,
                unsigned short* __restrict__ qB, unsigned short* __restrict__ oB) {
  int tid = blockIdx.x * blockDim.x + threadIdx.x;
  if (tid >= 131072) return;
  int which = tid >> 16;
  int e     = tid & 65535;
  const float* w = which ? out_w : q_w;
  unsigned short* dst = which ? oB : qB;
  int frag = e >> 9;
  int nt = frag >> 3, kt = frag & 7;
  int ln = (e >> 4) & 31;
  int p  = e & 15;
  int n  = nt * 16 + (ln & 15);
  int K  = kt * 32 + ((ln >> 4) << 3) + (p & 7) + ((p >> 3) << 4);
  dst[e] = f2bf(w[n * NC + K]);
}

// ===========================================================================
// Prototype K/V + pack into B fragments (single block)
// ===========================================================================
__global__ void __launch_bounds__(256)
pib_protokv_kernel(const float* __restrict__ proto_in,
                   const float* __restrict__ k_w, const float* __restrict__ k_b,
                   const float* __restrict__ v_w, const float* __restrict__ v_b,
                   unsigned short* __restrict__ kB, unsigned short* __restrict__ vB) {
  __shared__ float proto[NP * NC];
  __shared__ float kv[2][NP * NC];
  const int t = threadIdx.x;
  for (int i = t; i < NP * NC; i += 256) proto[i] = proto_in[i];
  __syncthreads();
  for (int task = t; task < 2 * NP * NC; task += 256) {
    int m = task >> 12;
    int rem = task & 4095;
    int p = rem >> 8, c = rem & 255;
    const float* wrow = (m ? v_w : k_w) + c * NC;
    float acc = m ? v_b[c] : k_b[c];
#pragma unroll 4
    for (int j = 0; j < NC; ++j) acc += proto[p * NC + j] * wrow[j];
    kv[m][p * NC + c] = acc;
  }
  __syncthreads();
  // kB: B[k=c][n=p] = k[p][c] ; 8 kt frags
  for (int e = t; e < 8 * 512; e += 256) {
    int kt = e >> 9;
    int ln = (e >> 4) & 31;
    int p  = e & 15;
    int n  = ln & 15;
    int K  = kt * 32 + ((ln >> 4) << 3) + (p & 7) + ((p >> 3) << 4);
    kB[e] = f2bf(kv[0][n * NC + K]);
  }
  // vB: B[k=p][n=c] = v[p][c], zero-padded K 16..31 ; 16 nt frags
  for (int e = t; e < 16 * 512; e += 256) {
    int nt = e >> 9;
    int ln = (e >> 4) & 31;
    int p  = e & 15;
    int K  = ((ln >> 4) << 3) + (p & 7) + ((p >> 3) << 4);
    float val = (K < NP) ? kv[1][K * NC + nt * 16 + (ln & 15)] : 0.0f;
    vB[e] = f2bf(val);
  }
}

// ===========================================================================
// Main fused attention kernel. 64 threads (2 waves), 32 pixels per block.
// ===========================================================================
__global__ void __launch_bounds__(64)
pib_main_kernel(const float* __restrict__ x, const float* __restrict__ pos,
                const float* __restrict__ q_b, const float* __restrict__ out_b,
                const unsigned short* __restrict__ qB, const unsigned short* __restrict__ oB,
                const unsigned short* __restrict__ kB, const unsigned short* __restrict__ vB,
                float* __restrict__ s) {
  __shared__ alignas(16) char smem[52224];
  unsigned short* xp   = (unsigned short*)smem;             // [32][264] bf16
  unsigned short* qs   = xp + 32 * 264;                     // [32][264] bf16
  unsigned short* outs = (unsigned short*)(smem + 33792);   // [32][264] bf16
  unsigned short* atts = (unsigned short*)(smem + 50688);   // [32][24]  bf16
  float* tbuf          = (float*)smem;                      // [2][16][260] f32 (aliases xp/qs)

  const int tid  = threadIdx.x;
  const int lane = tid & 31;
  const int wave = tid >> 5;
  const int pixBase = blockIdx.x * 32;
  const int b  = pixBase / NHW;
  const int hwBase = pixBase - b * NHW;

  // ---- Stage 1: xp = x + bilinear(pos_embed 64->96, half-pixel) ----
  {
    const int pix   = tid & 31;
    const int cHalf = tid >> 5;
    const int hw = hwBase + pix;
    const int h = hw / NW;
    const int w = hw - h * NW;
    const float fy = (h + 0.5f) * (2.0f / 3.0f) - 0.5f;
    const float fx = (w + 0.5f) * (2.0f / 3.0f) - 0.5f;
    int y0 = (int)floorf(fy), x0 = (int)floorf(fx);
    const float ty = fy - (float)y0;
    const float tx = fx - (float)x0;
    const int y0c = max(y0, 0), y1c = min(y0 + 1, 63);
    const int x0c = max(x0, 0), x1c = min(x0 + 1, 63);
    for (int i = 0; i < 128; ++i) {
      const int c = 2 * i + cHalf;
      const float* pc = pos + c * 4096;
      float p00 = pc[y0c * 64 + x0c], p01 = pc[y0c * 64 + x1c];
      float p10 = pc[y1c * 64 + x0c], p11 = pc[y1c * 64 + x1c];
      float ptop = p00 + (p01 - p00) * tx;
      float pbot = p10 + (p11 - p10) * tx;
      float pv = ptop + (pbot - ptop) * ty;
      float xv = x[(size_t)(b * NC + c) * NHW + hw];
      xp[pix * 264 + c] = f2bf(xv + pv);
    }
  }
  __syncthreads();

  const int rl   = wave * 16 + (lane & 15);  // A-fragment row (block-local)
  const int koff = (lane >> 4) << 3;         // 0 or 8 (A layout K split)
  const int hi8  = (lane >> 4) << 3;         // C/D layout row split
  const int colL = lane & 15;                // C/D layout column

  // ---- Stage 2: Q = xp * q_w^T + q_b ------------------------------------
  {
    FragU af[8];
#pragma unroll
    for (int kt = 0; kt < 8; ++kt) {
      af[kt].q[0] = *(const u32x4*)(xp + rl * 264 + kt * 32 + koff);
      af[kt].q[1] = *(const u32x4*)(xp + rl * 264 + kt * 32 + koff + 16);
    }
#pragma unroll
    for (int nt = 0; nt < 16; ++nt) {
      f32x8 acc = splat8(q_b[nt * 16 + colL]);
#pragma unroll
      for (int kt = 0; kt < 8; ++kt) {
        FragU bw;
        const u32x4* pw = (const u32x4*)(qB + ((nt * 8 + kt) << 9) + (lane << 4));
        bw.q[0] = pw[0]; bw.q[1] = pw[1];
        acc = __builtin_amdgcn_wmma_f32_16x16x32_bf16(false, af[kt].v, false, bw.v,
                                                      (short)0, acc, false, false);
      }
#pragma unroll
      for (int r = 0; r < 8; ++r)
        qs[(wave * 16 + r + hi8) * 264 + nt * 16 + colL] = f2bf(acc[r]);
    }
  }

  // ---- Stage 3: logits = Q * k^T (scale 1/16), softmax over P=16 --------
  {
    FragU aq[8];
#pragma unroll
    for (int kt = 0; kt < 8; ++kt) {
      aq[kt].q[0] = *(const u32x4*)(qs + rl * 264 + kt * 32 + koff);
      aq[kt].q[1] = *(const u32x4*)(qs + rl * 264 + kt * 32 + koff + 16);
    }
    f32x8 acc = splat8(0.0f);
#pragma unroll
    for (int kt = 0; kt < 8; ++kt) {
      FragU bw;
      const u32x4* pw = (const u32x4*)(kB + (kt << 9) + (lane << 4));
      bw.q[0] = pw[0]; bw.q[1] = pw[1];
      acc = __builtin_amdgcn_wmma_f32_16x16x32_bf16(false, aq[kt].v, false, bw.v,
                                                    (short)0, acc, false, false);
    }
#pragma unroll
    for (int r = 0; r < 8; ++r) {
      float lg = acc[r] * 0.0625f;   // C^-0.5 = 1/16
      float m = lg;
#pragma unroll
      for (int off = 1; off < 16; off <<= 1) m = fmaxf(m, __shfl_xor(m, off, 16));
      float e = expf(lg - m);
      float ssum = e;
#pragma unroll
      for (int off = 1; off < 16; off <<= 1) ssum += __shfl_xor(ssum, off, 16);
      atts[(wave * 16 + r + hi8) * 24 + colL] = f2bf(e / ssum);
    }
  }

  // ---- Stage 4: out = attn * v (K=16 zero-padded to 32) -----------------
  {
    FragU aat;
    aat.q[0] = *(const u32x4*)(atts + rl * 24 + koff);
    u32x4 z = {0u, 0u, 0u, 0u};
    aat.q[1] = z;
#pragma unroll
    for (int nt = 0; nt < 16; ++nt) {
      FragU bw;
      const u32x4* pw = (const u32x4*)(vB + (nt << 9) + (lane << 4));
      bw.q[0] = pw[0]; bw.q[1] = pw[1];
      f32x8 acc = __builtin_amdgcn_wmma_f32_16x16x32_bf16(false, aat.v, false, bw.v,
                                                          (short)0, splat8(0.0f), false, false);
#pragma unroll
      for (int r = 0; r < 8; ++r)
        outs[(wave * 16 + r + hi8) * 264 + nt * 16 + colL] = f2bf(acc[r]);
    }
  }

  // ---- Stage 5: s = out * out_w^T + out_b -------------------------------
  f32x8 accs[16];
  {
    FragU ao[8];
#pragma unroll
    for (int kt = 0; kt < 8; ++kt) {
      ao[kt].q[0] = *(const u32x4*)(outs + rl * 264 + kt * 32 + koff);
      ao[kt].q[1] = *(const u32x4*)(outs + rl * 264 + kt * 32 + koff + 16);
    }
#pragma unroll
    for (int nt = 0; nt < 16; ++nt) {
      f32x8 acc = splat8(out_b[nt * 16 + colL]);
#pragma unroll
      for (int kt = 0; kt < 8; ++kt) {
        FragU bw;
        const u32x4* pw = (const u32x4*)(oB + ((nt * 8 + kt) << 9) + (lane << 4));
        bw.q[0] = pw[0]; bw.q[1] = pw[1];
        acc = __builtin_amdgcn_wmma_f32_16x16x32_bf16(false, ao[kt].v, false, bw.v,
                                                      (short)0, acc, false, false);
      }
      accs[nt] = acc;
    }
  }

  // ---- Stage 6: LDS transpose -> coalesced NCHW store of s --------------
  __syncthreads();  // xp/qs regions dead block-wide; reuse as tbuf
  {
    float* tb = tbuf + wave * (16 * 260);
#pragma unroll
    for (int nt = 0; nt < 16; ++nt)
#pragma unroll
      for (int r = 0; r < 8; ++r)
        tb[(r + hi8) * 260 + nt * 16 + colL] = accs[nt][r];

    const int row  = lane & 15;
    const int dOff = lane >> 4;
    const int hwOut = hwBase + wave * 16 + row;
#pragma unroll 8
    for (int i = 0; i < 128; ++i) {
      const int d = 2 * i + dOff;
      s[(size_t)(b * NC + d) * NHW + hwOut] = tb[row * 260 + d];
    }
  }
}

// ===========================================================================
// Depthwise 3x3 + BN(eval) + exact GELU + residuals; deterministic partials
// ===========================================================================
__global__ void __launch_bounds__(256)
pib_local_kernel(const float* __restrict__ s, const float* __restrict__ x,
                 const float* __restrict__ dw_w,
                 const float* __restrict__ bn_g, const float* __restrict__ bn_b,
                 const float* __restrict__ bn_mean, const float* __restrict__ bn_var,
                 float* __restrict__ y, float* __restrict__ psum) {
  const int bc = blockIdx.x;          // b*256 + c
  const int c  = bc & 255;
  float w9[9];
#pragma unroll
  for (int i = 0; i < 9; ++i) w9[i] = dw_w[c * 9 + i];
  const float inv  = bn_g[c] * rsqrtf(bn_var[c] + 1e-5f);
  const float beta = bn_b[c] - bn_mean[c] * inv;
  const float* sp = s + (size_t)bc * NHW;
  const float* xr = x + (size_t)bc * NHW;
  float* yp = y + (size_t)bc * NHW;

  float lsum = 0.0f, lsq = 0.0f;
  for (int hw = threadIdx.x; hw < NHW; hw += 256) {
    const int h = hw / NW;
    const int w = hw - h * NW;
    float acc = 0.0f;
#pragma unroll
    for (int dh = -1; dh <= 1; ++dh) {
      const int nh = h + dh;
      if (nh < 0 || nh >= NH) continue;
#pragma unroll
      for (int dw = -1; dw <= 1; ++dw) {
        const int nw = w + dw;
        if (nw < 0 || nw >= NW) continue;
        acc += w9[(dh + 1) * 3 + (dw + 1)] * sp[nh * NW + nw];
      }
    }
    const float bn = acc * inv + beta;
    const float gel = 0.5f * bn * (1.0f + erff(bn * 0.70710678118654752f));
    const float yv = sp[hw] + gel + xr[hw];
    yp[hw] = yv;
    lsum += yv;
    lsq  += yv * yv;
  }
  __shared__ float r1[256], r2[256];
  r1[threadIdx.x] = lsum;
  r2[threadIdx.x] = lsq;
  __syncthreads();
  for (int off = 128; off > 0; off >>= 1) {
    if (threadIdx.x < off) {
      r1[threadIdx.x] += r1[threadIdx.x + off];
      r2[threadIdx.x] += r2[threadIdx.x + off];
    }
    __syncthreads();
  }
  if (threadIdx.x == 0) {
    psum[bc * 2 + 0] = r1[0];
    psum[bc * 2 + 1] = r2[0];
  }
}

// ===========================================================================
// GroupNorm(8) finalize
// ===========================================================================
__global__ void __launch_bounds__(256)
pib_gn_kernel(const float* __restrict__ y, const float* __restrict__ psum,
              const float* __restrict__ gn_g, const float* __restrict__ gn_b,
              float* __restrict__ out) {
  const int bc = blockIdx.x;
  const int b = bc >> 8;
  const int c = bc & 255;
  const int g = c >> 5;
  float s0 = 0.0f, s1 = 0.0f;
  const int base = (b * NC + (g << 5)) * 2;
#pragma unroll 4
  for (int j = 0; j < 32; ++j) {
    s0 += psum[base + 2 * j + 0];
    s1 += psum[base + 2 * j + 1];
  }
  const float invN = 1.0f / (32.0f * (float)NHW);
  const float mean = s0 * invN;
  const float var  = s1 * invN - mean * mean;
  const float rstd = rsqrtf(var + 1e-5f);
  const float ga = gn_g[c] * rstd;
  const float be = gn_b[c] - mean * ga;
  const float* yp = y + (size_t)bc * NHW;
  float* op = out + (size_t)bc * NHW;
  for (int hw = threadIdx.x; hw < NHW; hw += 256)
    op[hw] = yp[hw] * ga + be;
}

// ===========================================================================
extern "C" void kernel_launch(void* const* d_in, const int* in_sizes, int n_in,
                              void* d_out, int out_size, void* d_ws, size_t ws_size,
                              hipStream_t stream) {
  const float* x       = (const float*)d_in[0];
  const float* proto   = (const float*)d_in[1];
  const float* pos     = (const float*)d_in[2];
  const float* q_w     = (const float*)d_in[3];
  const float* q_b     = (const float*)d_in[4];
  const float* k_w     = (const float*)d_in[5];
  const float* k_b     = (const float*)d_in[6];
  const float* v_w     = (const float*)d_in[7];
  const float* v_b     = (const float*)d_in[8];
  const float* out_w   = (const float*)d_in[9];
  const float* out_b   = (const float*)d_in[10];
  const float* dw_w    = (const float*)d_in[11];
  const float* bn_g    = (const float*)d_in[12];
  const float* bn_b    = (const float*)d_in[13];
  const float* bn_mean = (const float*)d_in[14];
  const float* bn_var  = (const float*)d_in[15];
  const float* gn_g    = (const float*)d_in[16];
  const float* gn_b    = (const float*)d_in[17];
  float* outp = (float*)d_out;

  char* ws = (char*)d_ws;
  float*          psum = (float*)(ws + WS_PS);
  unsigned short* kB   = (unsigned short*)(ws + WS_KB);
  unsigned short* vB   = (unsigned short*)(ws + WS_VB);
  unsigned short* qB   = (unsigned short*)(ws + WS_QB);
  unsigned short* oB   = (unsigned short*)(ws + WS_OB);
  float*          s    = (float*)(ws + WS_S);
  float*          y    = (float*)(ws + WS_Y);

  pib_pack_kernel<<<512, 256, 0, stream>>>(q_w, out_w, qB, oB);
  pib_protokv_kernel<<<1, 256, 0, stream>>>(proto, k_w, k_b, v_w, v_b, kB, vB);
  pib_main_kernel<<<4608, 64, 0, stream>>>(x, pos, q_b, out_b, qB, oB, kB, vB, s);
  pib_local_kernel<<<NB * NC, 256, 0, stream>>>(s, x, dw_w, bn_g, bn_b, bn_mean,
                                                bn_var, y, psum);
  pib_gn_kernel<<<NB * NC, 256, 0, stream>>>(y, psum, gn_g, gn_b, outp);
}
